// AutoregBlock_66709432041661
// MI455X (gfx1250) — matrix operand order repaired and checked by reading, not compiled, not run on hardware
//
#include <hip/hip_runtime.h>
#include <cstdint>

typedef __attribute__((ext_vector_type(16))) _Float16 v16h;
typedef __attribute__((ext_vector_type(8)))  _Float16 v8h;
typedef __attribute__((ext_vector_type(8)))  float    v8f;

#define NB   1024
#define DIM  384
#define HID  64
#define KK   8
#define G    48
#define MID  128
#define NT   512      // 16 waves of 32
#define NWAVE (NT/32)

// ---- LDS layout (bytes) ----
#define OFF_H    0          // H    f16 384x64   = 49152 B
#define OFF_T    49152      // T    f16 48x512 (later reused as Hs 384x64) = 49152 B
#define OFF_Z1   98304      // z1   f16 384x128  = 98304 B
#define OFF_DW   196608     // down_w f16 128x64 = 16384 B
#define OFF_UW   212992     // up_w   f16 64x128 = 16384 B
#define OFF_PG   229376     // Pg   f32 48x48    = 9216 B
#define OFF_WSM  238592     // softmax(patch) f32 64 = 256 B
#define OFF_SCL  238848     // scales f32 384    = 1536 B
#define OFF_RW   240384     // row_weights f32 48 = 192 B
#define SMEM_BYTES 240576

__device__ __forceinline__ float leakyf(float v) { return v >= 0.0f ? v : 0.01f * v; }

// A fragment: 16x32 f16, M = lane&15, K quarters interleaved by lane-half.
// f[0..7]  (VGPR0-3) = K = half*8 + 0..7   (contiguous)
// f[8..15] (VGPR4-7) = K = 16 + half*8 + 0..7
__device__ __forceinline__ v16h load_a_frag(const _Float16* base, int stride, int lane) {
    const _Float16* p = base + (lane & 15) * stride + ((lane >> 4) << 3);
    v8h lo = *(const v8h*)(p);
    v8h hi = *(const v8h*)(p + 16);
    v16h r;
#pragma unroll
    for (int i = 0; i < 8; ++i) { r[i] = lo[i]; r[i + 8] = hi[i]; }
    return r;
}

// B fragment: 32x16 f16 loaded from row-major (N x K) buffer (computes X*Y^T).
// N = lane&15, f[0..7] = K = half*8 + 0..7, f[8..15] = K = 16 + half*8 + 0..7 (matches A).
__device__ __forceinline__ v16h load_b_frag(const _Float16* base, int stride, int lane) {
    const _Float16* p = base + (lane & 15) * stride + ((lane >> 4) << 3);
    v8h lo = *(const v8h*)(p);
    v8h hi = *(const v8h*)(p + 16);
    v16h r;
#pragma unroll
    for (int i = 0; i < 8; ++i) { r[i] = lo[i]; r[i + 8] = hi[i]; }
    return r;
}

__global__ void __launch_bounds__(NT, 1)
fused_block_kernel(const float* __restrict__ x,       // (B,DIM)
                   const float* __restrict__ Wmat,    // (DIM,HID)
                   const float* __restrict__ bmat,    // (DIM,HID)
                   const float* __restrict__ patchw,  // (8,8)
                   const float* __restrict__ pw1, const float* __restrict__ pb1,
                   const float* __restrict__ pw2, const float* __restrict__ pb2,
                   const float* __restrict__ projw,   // (DIM,G)
                   const float* __restrict__ projb,   // (DIM)
                   const float* __restrict__ downw,   // (MID,HID)
                   const float* __restrict__ downb,   // (MID)
                   const float* __restrict__ upw,     // (HID,MID)
                   const float* __restrict__ upb,     // (HID)
                   float* __restrict__ out)           // (B,DIM,HID)
{
    extern __shared__ char smem[];
    _Float16* Hs16 = (_Float16*)(smem + OFF_H);    // H (384x64)
    _Float16* Ts16 = (_Float16*)(smem + OFF_T);    // T (48x512) / Hscaled (384x64)
    _Float16* Z1   = (_Float16*)(smem + OFF_Z1);   // z1 (384x128)
    _Float16* DW   = (_Float16*)(smem + OFF_DW);   // down_w (128x64)
    _Float16* UW   = (_Float16*)(smem + OFF_UW);   // up_w (64x128)
    float* PG  = (float*)(smem + OFF_PG);
    float* WSM = (float*)(smem + OFF_WSM);
    float* SCL = (float*)(smem + OFF_SCL);
    float* RW  = (float*)(smem + OFF_RW);

    const int tid  = threadIdx.x;
    const int lane = tid & 31;
    const int wave = tid >> 5;
    const int bb   = blockIdx.x;

    // ---------------- P0: stage weights (f16), patch softmax, H = gelu(s*W + inv*b)
    if (tid == 0) {
        float m = patchw[0];
        for (int i = 1; i < 64; ++i) m = fmaxf(m, patchw[i]);
        float s = 0.0f;
        for (int i = 0; i < 64; ++i) { float e = expf(patchw[i] - m); WSM[i] = e; s += e; }
        float inv = 1.0f / s;
        for (int i = 0; i < 64; ++i) WSM[i] *= inv;
    }
    for (int e = tid; e < MID * HID; e += NT) DW[e] = (_Float16)downw[e];
    for (int e = tid; e < HID * MID; e += NT) UW[e] = (_Float16)upw[e];
    {
        const float* xb = x + (size_t)bb * DIM;
        for (int e = tid; e < DIM * HID; e += NT) {
            int row = e >> 6;
            float s  = xb[row];
            float iv = 1.0f / (fabsf(s) + 1e-6f);
            float v  = s * Wmat[e] + iv * bmat[e];
            float g  = 0.5f * v * (1.0f + erff(v * 0.70710678118654752f)); // exact gelu
            Hs16[e]  = (_Float16)g;
        }
    }
    __syncthreads();

    // ---------------- P1: T[g, l*64+c] = sum_k w[k,l] * H[8g+k, c]   (48 x 512)
    for (int e = tid; e < G * 512; e += NT) {
        int g = e >> 9;
        int r = e & 511;
        int l = r >> 6;
        int c = r & 63;
        float acc = 0.0f;
#pragma unroll
        for (int k = 0; k < KK; ++k)
            acc += WSM[k * 8 + l] * (float)Hs16[(((g << 3) + k) << 6) + c];
        Ts16[e] = (_Float16)acc;
    }
    __syncthreads();

    // ---------------- P2: Pg(48x48) = T(48x512) @ H(48x512)^T  via WMMA
    for (int t = wave; t < 9; t += NWAVE) {
        int tr = t / 3, tc = t % 3;
        v8f acc = {};
#pragma unroll
        for (int kc = 0; kc < 512; kc += 32) {
            v16h a = load_a_frag(Ts16 + (tr * 16) * 512 + kc, 512, lane);
            v16h bfr = load_b_frag(Hs16 + (tc * 16) * 512 + kc, 512, lane);
            acc = __builtin_amdgcn_wmma_f32_16x16x32_f16(false, a, false, bfr,
                                                         (short)0, acc, false, false);
        }
        int n  = lane & 15;
        int mb = (lane >> 4) << 3;
#pragma unroll
        for (int r = 0; r < 8; ++r)
            PG[(tr * 16 + mb + r) * G + tc * 16 + n] = acc[r];
    }
    __syncthreads();

    // ---------------- P3: two leaky elementwise pair transforms
    for (int e = tid; e < G * G; e += NT) {
        float v = PG[e];
        v = leakyf(v * pw1[e] + pb1[e]);
        v = leakyf(v * pw2[e] + pb2[e]);
        PG[e] = v;
    }
    __syncthreads();

    // ---------------- P4: row softmax -> row_weights[g] = sum 1/(1+p^2)
    if (tid < G) {
        const float* row = PG + tid * G;
        float m = row[0];
        for (int h = 1; h < G; ++h) m = fmaxf(m, row[h]);
        float s = 0.0f;
        for (int h = 0; h < G; ++h) s += expf(row[h] - m);
        float invs = 1.0f / s;
        float rw = 0.0f;
        for (int h = 0; h < G; ++h) {
            float p = expf(row[h] - m) * invs;
            rw += 1.0f / (1.0f + p * p);
        }
        RW[tid] = rw;
    }
    __syncthreads();

    // ---------------- P5: scales = rw @ proj_w^T + proj_b   (384)
    for (int i = tid; i < DIM; i += NT) {
        float acc = projb[i];
        const float* pw = projw + i * G;
#pragma unroll 8
        for (int g = 0; g < G; ++g) acc += RW[g] * pw[g];
        SCL[i] = acc;
    }
    __syncthreads();

    // ---------------- P6: Hscaled (into T buffer, viewed 384x64)
    for (int e = tid; e < DIM * HID; e += NT) {
        int row = e >> 6;
        Ts16[e] = (_Float16)((float)Hs16[e] * SCL[row]);
    }
    __syncthreads();

    // ---------------- P7: z1(384x128) = leaky(Hs @ down_w^T + down_b) via WMMA
    for (int t = wave; t < 24 * 8; t += NWAVE) {
        int tr = t >> 3, tc = t & 7;
        v8f acc = {};
#pragma unroll
        for (int kc = 0; kc < HID; kc += 32) {
            v16h a   = load_a_frag(Ts16 + (tr * 16) * HID + kc, HID, lane);
            v16h bfr = load_b_frag(DW + (tc * 16) * HID + kc, HID, lane);
            acc = __builtin_amdgcn_wmma_f32_16x16x32_f16(false, a, false, bfr,
                                                         (short)0, acc, false, false);
        }
        int n  = lane & 15;
        int mb = (lane >> 4) << 3;
        float bias = downb[tc * 16 + n];
#pragma unroll
        for (int r = 0; r < 8; ++r) {
            float v = leakyf(acc[r] + bias);
            Z1[(tr * 16 + mb + r) * MID + tc * 16 + n] = (_Float16)v;
        }
    }
    __syncthreads();

    // ---------------- P8: z(384x64) = leaky(z1 @ up_w^T + up_b) -> global (f32)
    float* outb = out + (size_t)bb * DIM * HID;
    for (int t = wave; t < 24 * 4; t += NWAVE) {
        int tr = t >> 2, tc = t & 3;
        v8f acc = {};
#pragma unroll
        for (int kc = 0; kc < MID; kc += 32) {
            v16h a   = load_a_frag(Z1 + (tr * 16) * MID + kc, MID, lane);
            v16h bfr = load_b_frag(UW + (tc * 16) * MID + kc, MID, lane);
            acc = __builtin_amdgcn_wmma_f32_16x16x32_f16(false, a, false, bfr,
                                                         (short)0, acc, false, false);
        }
        int n  = lane & 15;
        int mb = (lane >> 4) << 3;
        float bias = upb[tc * 16 + n];
#pragma unroll
        for (int r = 0; r < 8; ++r) {
            float v = leakyf(acc[r] + bias);
            outb[(tr * 16 + mb + r) * HID + tc * 16 + n] = v;
        }
    }
}

extern "C" void kernel_launch(void* const* d_in, const int* in_sizes, int n_in,
                              void* d_out, int out_size, void* d_ws, size_t ws_size,
                              hipStream_t stream) {
    (void)in_sizes; (void)n_in; (void)out_size; (void)d_ws; (void)ws_size;
    static_assert(SMEM_BYTES <= 320 * 1024, "fits WGP LDS");
    (void)hipFuncSetAttribute((const void*)fused_block_kernel,
                              hipFuncAttributeMaxDynamicSharedMemorySize, SMEM_BYTES);
    fused_block_kernel<<<NB, NT, SMEM_BYTES, stream>>>(
        (const float*)d_in[0],  (const float*)d_in[1],  (const float*)d_in[2],
        (const float*)d_in[3],  (const float*)d_in[4],  (const float*)d_in[5],
        (const float*)d_in[6],  (const float*)d_in[7],  (const float*)d_in[8],
        (const float*)d_in[9],  (const float*)d_in[10], (const float*)d_in[11],
        (const float*)d_in[12], (const float*)d_in[13],
        (float*)d_out);
}
